// CFConv_46342697124299
// MI455X (gfx1250) — matrix operand order repaired and checked by reading, not compiled, not run on hardware
//
#include <hip/hip_runtime.h>
#include <hip/hip_bf16.h>

// CFConv for MI455X (gfx1250, wave32, WMMA).
//   weight = relu(rbf@W1+b1)@W2+b2 ; h = x@Wl+bl ; out = segment_sum(h[col]*weight, row)
// GEMMs on v_wmma_f32_16x16x32_f16 (f16 inputs, fp32 accumulate).
// Workspace requirement: N_NODES * OUT_CH * 4 bytes (~20.5 MB) for h.

typedef __attribute__((ext_vector_type(16))) _Float16 v16h;
typedef __attribute__((ext_vector_type(8)))  _Float16 v8h;
typedef __attribute__((ext_vector_type(4)))  _Float16 v4h;
typedef __attribute__((ext_vector_type(8)))  float    v8f;
typedef __attribute__((ext_vector_type(4)))  float    v4f;

#define IN_CH   128
#define OUT_CH  128
#define NUM_RBF 64
// LDS strides (in halves), padded +8 to spread the 64 LDS banks.
#define LDA1 72    // rbf tile rows (K=64)
#define LDB1 72    // W1 columns   (K=64)
#define LDK  136   // x/hidden rows and W2/Wl columns (K=128)

#define EDGE_GRID_MAX 1280   // persistent edge blocks (tiles are grid-strided)

static __device__ __forceinline__ v16h cat8(v8h lo, v8h hi) {
  return __builtin_shufflevector(lo, hi, 0,1,2,3,4,5,6,7,8,9,10,11,12,13,14,15);
}

// A-matrix fragment (16x32 f16): lane row = lane%16; K runs [kb+8*hs, +8) and [kb+8*hs+16, +8)
static __device__ __forceinline__ v16h load_fragA(const _Float16* rowp, int kb, int hs) {
  const _Float16* p = rowp + kb + 8 * hs;
  v8h lo = *(const v8h*)p;
  v8h hi = *(const v8h*)(p + 16);
  return cat8(lo, hi);
}
// B-matrix fragment (32x16 f16): lane col = lane%16; K runs [kb+8*hs, +8) and [kb+8*hs+16, +8)
static __device__ __forceinline__ v16h load_fragB(const _Float16* colp, int kb, int hs) {
  const _Float16* p = colp + kb + 8 * hs;
  v8h lo = *(const v8h*)p;
  v8h hi = *(const v8h*)(p + 16);
  return cat8(lo, hi);
}

// Native no-return fp32 atomic add with compile-time instruction offset.
static __device__ __forceinline__ void atomic_add_f32_off(float* p, float v, int byte_off) {
  asm volatile("global_atomic_add_f32 %0, %1, off offset:%c2"
               :: "v"(p), "v"(v), "i"(byte_off) : "memory");
}
static __device__ __forceinline__ void atomic_add_f32(float* p, float v) {
  asm volatile("global_atomic_add_f32 %0, %1, off" :: "v"(p), "v"(v) : "memory");
}

__global__ void __launch_bounds__(256) cfconv_zero_kernel(float* __restrict__ out, int n) {
  int i = (blockIdx.x * blockDim.x + threadIdx.x) * 4;
  v4f z = {0.f, 0.f, 0.f, 0.f};
  if (i + 3 < n) {
    *(v4f*)(out + i) = z;
  } else {
    for (int j = i; j < n; ++j) out[j] = 0.f;
  }
}

// h = x @ Wl + bl   (per block: 128 nodes, 8 waves x 16-node tiles)
__global__ void __launch_bounds__(256) cfconv_node_linear_kernel(
    const float* __restrict__ x, const float* __restrict__ Wl,
    const float* __restrict__ bl, float* __restrict__ h, int nNodes) {
  extern __shared__ _Float16 smemN[];
  _Float16* ldsW = smemN;                 // [OUT_CH cols][LDK]  (col-major, K-contig)
  _Float16* ldsX = smemN + OUT_CH * LDK;  // [128 rows][LDK]     (row-major)

  const int tid = threadIdx.x;
  const int nb  = blockIdx.x * 128;
  const bool fullTile = (nb + 128 <= nNodes);

  // Stage Wl (fp32 [K=128][N=128] row-major) -> f16 B-layout
  #pragma unroll 4
  for (int idx = tid; idx < IN_CH * OUT_CH; idx += 256) {
    int k = idx >> 7, n = idx & 127;
    ldsW[n * LDK + k] = (_Float16)Wl[idx];
  }
  // Stage x tile -> f16 A-layout (zero-pad past nNodes)
  #pragma unroll 4
  for (int q = tid; q < 128 * (IN_CH / 4); q += 256) {
    int r = q >> 5, c = (q & 31) * 4;
    int node = nb + r;
    v4f xv = {0.f, 0.f, 0.f, 0.f};
    if (node < nNodes) xv = *(const v4f*)(x + (size_t)node * IN_CH + c);
    v4h xh;
    #pragma unroll
    for (int i = 0; i < 4; ++i) xh[i] = (_Float16)xv[i];
    *(v4h*)(ldsX + r * LDK + c) = xh;
  }
  __syncthreads();

  const int lane = tid & 31, wave = tid >> 5;
  const int hs = lane >> 4, l16 = lane & 15;
  const _Float16* arow = ldsX + (wave * 16 + l16) * LDK;

  v16h afrag[4];
  #pragma unroll
  for (int kt = 0; kt < 4; ++kt) afrag[kt] = load_fragA(arow, kt * 32, hs);

  const int m_base = nb + wave * 16 + 8 * hs;  // global node of acc[0]
  #pragma unroll
  for (int nt = 0; nt < 8; ++nt) {
    const int n = nt * 16 + l16;
    float bv = bl[n];
    v8f acc;
    #pragma unroll
    for (int i = 0; i < 8; ++i) acc[i] = bv;
    #pragma unroll
    for (int kt = 0; kt < 4; ++kt) {
      v16h bfrag = load_fragB(ldsW + n * LDK, kt * 32, hs);
      acc = __builtin_amdgcn_wmma_f32_16x16x32_f16(false, afrag[kt], false, bfrag,
                                                   (short)0, acc, false, false);
    }
    float* hp = h + (size_t)m_base * OUT_CH + n;   // stride OUT_CH per v -> imm offsets
    if (fullTile) {
      #pragma unroll
      for (int v = 0; v < 8; ++v) hp[v * OUT_CH] = acc[v];
    } else {
      #pragma unroll
      for (int v = 0; v < 8; ++v)
        if (m_base + v < nNodes) hp[v * OUT_CH] = acc[v];
    }
  }
}

// Persistent fused kernel: filter-net (2 WMMA layers) + gather h[col] + atomic scatter.
// Each block stages W1/W2 once, then grid-strides over 128-edge tiles.
__global__ void __launch_bounds__(256) cfconv_edge_kernel(
    const long long* __restrict__ edge_index, const float* __restrict__ rbf,
    const float* __restrict__ W1, const float* __restrict__ b1,
    const float* __restrict__ W2, const float* __restrict__ b2,
    const float* __restrict__ h, float* __restrict__ out,
    int nEdges, int nTiles) {
  extern __shared__ char smemE[];
  _Float16* ldsW1 = (_Float16*)smemE;             // [128 cols][LDB1]
  _Float16* ldsW2 = ldsW1 + OUT_CH * LDB1;        // [128 cols][LDK]
  _Float16* ldsR  = ldsW2 + OUT_CH * LDK;         // [128 edges][LDA1]
  _Float16* ldsH  = ldsR  + 128 * LDA1;           // [128 edges][LDK]
  int* ldsRow = (int*)(ldsH + 128 * LDK);
  int* ldsCol = ldsRow + 128;

  const int tid = threadIdx.x;
  const int lane = tid & 31, wave = tid >> 5;
  const int hs = lane >> 4, l16 = lane & 15;

  // Stage W1 [64][128] and W2 [128][128] -> f16 B-layout (once per persistent block)
  #pragma unroll 4
  for (int idx = tid; idx < NUM_RBF * OUT_CH; idx += 256) {
    int k = idx >> 7, n = idx & 127;
    ldsW1[n * LDB1 + k] = (_Float16)W1[idx];
  }
  #pragma unroll 4
  for (int idx = tid; idx < OUT_CH * OUT_CH; idx += 256) {
    int k = idx >> 7, n = idx & 127;
    ldsW2[n * LDK + k] = (_Float16)W2[idx];
  }

  for (int tile = blockIdx.x; tile < nTiles; tile += gridDim.x) {
    const int eb = tile * 128;
    const bool fullTile = (eb + 128 <= nEdges);

    // Barrier: weights ready (first iter) / previous tile's ldsR+indices consumed (later)
    __syncthreads();

    // Stage rbf tile [128 edges][64] -> f16 A-layout (coalesced b128 global loads)
    #pragma unroll 4
    for (int q = tid; q < 128 * (NUM_RBF / 4); q += 256) {
      int r = q >> 4, c = (q & 15) * 4;
      v4f xv = {0.f, 0.f, 0.f, 0.f};
      if (eb + r < nEdges) xv = *(const v4f*)(rbf + (size_t)(eb + r) * NUM_RBF + c);
      v4h xh;
      #pragma unroll
      for (int i = 0; i < 4; ++i) xh[i] = (_Float16)xv[i];
      *(v4h*)(ldsR + r * LDA1 + c) = xh;
    }
    // Stage edge indices (int64 -> int32)
    if (tid < 128) {
      int e = eb + tid;
      ldsRow[tid] = (e < nEdges) ? (int)edge_index[e] : 0;
      ldsCol[tid] = (e < nEdges) ? (int)edge_index[(size_t)nEdges + e] : 0;
    }
    __syncthreads();

    // Prefetch next tile's rbf stream (one 128B line per thread) while we compute.
    {
      int nxt = tile + gridDim.x;
      if (nxt < nTiles) {
        const char* pf = (const char*)(rbf + (size_t)nxt * 128 * NUM_RBF) + tid * 128;
        __builtin_prefetch(pf, 0, 3);
      }
    }

    // ---- Layer 1: hidden = relu(rbf @ W1 + b1); restage into LDS A-layout.
    // Each wave only touches its own 16 rows of ldsH (DS ops in-order per wave).
    v16h a1[2];
    const _Float16* r1 = ldsR + (wave * 16 + l16) * LDA1;
    #pragma unroll
    for (int kt = 0; kt < 2; ++kt) a1[kt] = load_fragA(r1, kt * 32, hs);

    #pragma unroll
    for (int nt = 0; nt < 8; ++nt) {
      const int n = nt * 16 + l16;
      float bv = b1[n];
      v8f acc;
      #pragma unroll
      for (int i = 0; i < 8; ++i) acc[i] = bv;
      #pragma unroll
      for (int kt = 0; kt < 2; ++kt) {
        v16h bfrag = load_fragB(ldsW1 + n * LDB1, kt * 32, hs);
        acc = __builtin_amdgcn_wmma_f32_16x16x32_f16(false, a1[kt], false, bfrag,
                                                     (short)0, acc, false, false);
      }
      #pragma unroll
      for (int v = 0; v < 8; ++v) {
        float r = acc[v] > 0.f ? acc[v] : 0.f;     // ReLU
        int m = wave * 16 + v + 8 * hs;            // C/D layout: M = v + 8*(lane>=16)
        ldsH[m * LDK + n] = (_Float16)r;           // hidden element (edge m, k = n)
      }
    }

    // ---- Layer 2 + fused gather/modulate/scatter
    const float* hp[8];
    float*       op[8];
    #pragma unroll
    for (int v = 0; v < 8; ++v) {
      int eloc = wave * 16 + v + 8 * hs;
      hp[v] = h   + (size_t)ldsCol[eloc] * OUT_CH + l16;
      op[v] = out + (size_t)ldsRow[eloc] * OUT_CH + l16;
    }
    v16h a2[4];
    const _Float16* r2 = ldsH + (wave * 16 + l16) * LDK;
    #pragma unroll
    for (int kt = 0; kt < 4; ++kt) a2[kt] = load_fragA(r2, kt * 32, hs);

    #pragma unroll
    for (int nt = 0; nt < 8; ++nt) {
      const int n = nt * 16 + l16;
      float bv = b2[n];
      v8f acc;
      #pragma unroll
      for (int i = 0; i < 8; ++i) acc[i] = bv;
      #pragma unroll
      for (int kt = 0; kt < 4; ++kt) {
        v16h bfrag = load_fragB(ldsW2 + n * LDK, kt * 32, hs);
        acc = __builtin_amdgcn_wmma_f32_16x16x32_f16(false, a2[kt], false, bfrag,
                                                     (short)0, acc, false, false);
      }
      if (fullTile) {
        #pragma unroll
        for (int v = 0; v < 8; ++v) {
          float hv = hp[v][nt * 16];                          // imm-offset L2 gather
          atomic_add_f32_off(op[v], acc[v] * hv, nt * 64);    // imm-offset scatter
        }
      } else {
        #pragma unroll
        for (int v = 0; v < 8; ++v) {
          int eloc = wave * 16 + v + 8 * hs;
          if (eb + eloc < nEdges) {
            float hv = hp[v][nt * 16];
            atomic_add_f32((float*)(op[v] + nt * 16), acc[v] * hv);
          }
        }
      }
    }
  }
}

extern "C" void kernel_launch(void* const* d_in, const int* in_sizes, int n_in,
                              void* d_out, int out_size, void* d_ws, size_t ws_size,
                              hipStream_t stream) {
  const float*     x   = (const float*)d_in[0];
  const long long* ei  = (const long long*)d_in[1];
  const float*     rbf = (const float*)d_in[2];
  const float*     W1  = (const float*)d_in[3];
  const float*     b1  = (const float*)d_in[4];
  const float*     W2  = (const float*)d_in[5];
  const float*     b2  = (const float*)d_in[6];
  const float*     Wl  = (const float*)d_in[7];
  const float*     bl  = (const float*)d_in[8];
  float* out = (float*)d_out;
  float* h   = (float*)d_ws;   // [nNodes][OUT_CH] fp32

  const int nNodes = in_sizes[0] / IN_CH;
  const int nEdges = in_sizes[1] / 2;

  // 1) zero the accumulation target
  int zthreads = (out_size + 3) / 4;
  cfconv_zero_kernel<<<(zthreads + 255) / 256, 256, 0, stream>>>(out, out_size);

  // 2) node projection h = x@Wl + bl
  size_t smemN = (size_t)(OUT_CH * LDK + 128 * LDK) * sizeof(_Float16);
  cfconv_node_linear_kernel<<<(nNodes + 127) / 128, 256, smemN, stream>>>(x, Wl, bl, h, nNodes);

  // 3) persistent fused filter-net + gather + scatter
  const int nTiles = (nEdges + 127) / 128;
  const int grid   = nTiles < EDGE_GRID_MAX ? nTiles : EDGE_GRID_MAX;
  size_t smemEb = (size_t)(OUT_CH * LDB1 + OUT_CH * LDK + 128 * LDA1 + 128 * LDK) * sizeof(_Float16)
                + 256 * sizeof(int);
  cfconv_edge_kernel<<<grid, 256, smemEb, stream>>>(ei, rbf, W1, b1, W2, b2, h, out,
                                                    nEdges, nTiles);
}
